// GeometricMedianGCN_40973988003892
// MI455X (gfx1250) — matrix operand order repaired and checked by reading, not compiled, not run on hardware
//
#include <hip/hip_runtime.h>

// GeometricMedianGCN for MI455X (gfx1250, wave32).
// GEMMs via v_wmma_f32_16x16x32_f16 with fully-unrolled, guard-free K loops
// (operands zero-padded in workspace). Graph Weiszfeld phase via channel-
// parallel edge waves + f32 global atomics (node arrays are L2-resident:
// ~70MB << 192MB L2 on MI455X).

#define EPSV  1e-5f
#define EPS2V 1e-10f

typedef __attribute__((ext_vector_type(16))) _Float16 v16h;
typedef __attribute__((ext_vector_type(8)))  float    v8f;

// ---------------------------------------------------------------- helpers ----
template <int W>
static __device__ __forceinline__ float grp_sum(float v) {
#pragma unroll
  for (int off = W / 2; off > 0; off >>= 1) v += __shfl_xor(v, off, 32);
  return v;
}

static __device__ __forceinline__ float grp_max32(float v) {
#pragma unroll
  for (int off = 16; off > 0; off >>= 1) v = fmaxf(v, __shfl_xor(v, off, 32));
  return v;
}

// Unguarded: 8 consecutive floats -> 8 fp16 (two b128 loads, v_cvt_pk_f16_f32).
static __device__ __forceinline__ void load8_h(const float* __restrict__ p, _Float16* o) {
  const float4* q = reinterpret_cast<const float4*>(p);
  float4 a = q[0], b = q[1];
  o[0] = (_Float16)a.x; o[1] = (_Float16)a.y; o[2] = (_Float16)a.z; o[3] = (_Float16)a.w;
  o[4] = (_Float16)b.x; o[5] = (_Float16)b.y; o[6] = (_Float16)b.z; o[7] = (_Float16)b.w;
}

// ------------------------------------------------------------------ GEMM -----
// Cout[M,Nstore] = A[M,K] @ B[Nb,K]^T + bias  (B/bias pre-padded to tile size;
// K a multiple of 32 at compile time -> fully unrolled, branch-free hot loop).
template <int K>
__global__ void gemm_wmma_f16(const float* __restrict__ A, const float* __restrict__ B,
                              const float* __restrict__ bias, float* __restrict__ Cout,
                              int M, int Nstore) {
  const int wave  = threadIdx.x >> 5;
  const int lane  = threadIdx.x & 31;
  const int tileM = blockIdx.x * (blockDim.x >> 5) + wave;
  const int tileN = blockIdx.y;
  if (tileM * 16 >= M) return;                 // wave-uniform exit

  const int half = lane >> 4;
  const int l16  = lane & 15;
  const int arow = min(tileM * 16 + l16, M - 1);  // branch-free clamp
  const int brow = tileN * 16 + l16;              // B padded: always valid
  const float* ap = A + (size_t)arow * K;
  const float* bp = B + (size_t)brow * K;

  v8f acc = {};
#pragma unroll
  for (int kb = 0; kb < K / 32; ++kb) {
    const int kbase = kb * 32;
    _Float16 ab[16], bb[16];
    // A k order chosen here (matches B): half0 lanes hold K {0..15}, half1 {16..31} (contiguous)
    load8_h(ap + kbase + half * 16,     ab);
    load8_h(ap + kbase + half * 16 + 8, ab + 8);
    // B k order chosen here (matches A): half0 lanes hold K {0..15}, half1 {16..31} (contiguous)
    load8_h(bp + kbase + half * 16,     bb);
    load8_h(bp + kbase + half * 16 + 8, bb + 8);
    v16h av, bv;
#pragma unroll
    for (int i = 0; i < 16; ++i) { av[i] = ab[i]; bv[i] = bb[i]; }
    acc = __builtin_amdgcn_wmma_f32_16x16x32_f16(false, av, false, bv,
                                                 (short)0, acc, false, false);
  }
  // D layout: VGPR r -> M = half*8 + r; N = lane&15
#pragma unroll
  for (int r = 0; r < 8; ++r) {
    int mm = tileM * 16 + half * 8 + r;
    int nn = tileN * 16 + l16;
    if (mm < M && nn < Nstore) Cout[(size_t)mm * Nstore + nn] = acc[r] + bias[nn];
  }
}

// Zero-pad W2[40,16] -> W2p[48,32] and b2[40] -> b2p[48].
__global__ void pad_w2_k(const float* __restrict__ W2, const float* __restrict__ b2,
                         float* __restrict__ W2p, float* __restrict__ b2p) {
  int gid = blockIdx.x * blockDim.x + threadIdx.x;
  if (gid < 48 * 32) {
    int r = gid / 32, c = gid % 32;
    W2p[gid] = (r < 40 && c < 16) ? W2[r * 16 + c] : 0.0f;
  }
  if (gid < 48) b2p[gid] = (gid < 40) ? b2[gid] : 0.0f;
}

// relu(med1)[N,16] -> zero-padded [N,32] for the K=32 layer-2 GEMM.
__global__ void relu_pad_k(const float* __restrict__ med, float* __restrict__ hp, int Nn) {
  int gid = blockIdx.x * blockDim.x + threadIdx.x;
  if (gid >= Nn * 32) return;
  int i = gid >> 5, c = gid & 31;
  hp[gid] = (c < 16) ? fmaxf(med[i * 16 + c], 0.0f) : 0.0f;
}

// --------------------------------------------------------- Weiszfeld phase --
// C==16 -> 16 lanes/edge (1 ch each). C==40 -> 32 lanes/edge (lanes 0..7 carry 2).

template <int C>
__global__ void mean_init_k(const float* __restrict__ h, float* __restrict__ acc,
                            float* __restrict__ cnt, int Nn) {
  int gid = blockIdx.x * blockDim.x + threadIdx.x;
  if (gid >= Nn * C) return;
  acc[gid] = h[gid];                 // self-loop message
  if (gid % C == 0) cnt[gid / C] = 1.0f;
}

template <int C>
__global__ void mean_edge_k(const float* __restrict__ h, const int* __restrict__ src,
                            const int* __restrict__ dst, float* __restrict__ acc,
                            float* __restrict__ cnt, int E) {
  constexpr int LPE = (C <= 16) ? 16 : 32;
  int gid = blockIdx.x * blockDim.x + threadIdx.x;
  int e = gid / LPE, lane = gid % LPE;
  if (e >= E) return;
  int s = src[e], t = dst[e];
  atomicAdd(acc + t * C + lane, h[s * C + lane]);
  if (C > 32 && lane < C - 32) atomicAdd(acc + t * C + lane + 32, h[s * C + lane + 32]);
  if (lane == 0) atomicAdd(cnt + t, 1.0f);
}

template <int C>
__global__ void mean_final_k(const float* __restrict__ acc, const float* __restrict__ cnt,
                             float* __restrict__ med, int* __restrict__ act, int Nn) {
  int gid = blockIdx.x * blockDim.x + threadIdx.x;
  if (gid >= Nn * C) return;
  int i = gid / C;
  med[gid] = acc[gid] / cnt[i];
  if (gid % C == 0) act[i] = 1;
}

// Re-initialize ws/wt each iteration with the self-loop term (no memset needed).
template <int C>
__global__ void self_init_k(const float* __restrict__ h, const float* __restrict__ med,
                            float* __restrict__ ws, float* __restrict__ wt, int Nn) {
  constexpr int LPE = (C <= 16) ? 16 : 32;
  int gid = blockIdx.x * blockDim.x + threadIdx.x;
  int i = gid / LPE, lane = gid % LPE;
  if (i >= Nn) return;
  float m0 = h[i * C + lane], mu0 = med[i * C + lane];
  float d0 = m0 - mu0, sq = d0 * d0;
  float m1 = 0.0f;
  if (C > 32 && lane < C - 32) {
    m1 = h[i * C + lane + 32];
    float d1 = m1 - med[i * C + lane + 32];
    sq += d1 * d1;
  }
  sq = grp_sum<LPE>(sq);
  float inv = 1.0f / sqrtf(fmaxf(sq, EPS2V));
  ws[i * C + lane] = m0 * inv;
  if (C > 32 && lane < C - 32) ws[i * C + lane + 32] = m1 * inv;
  if (lane == 0) wt[i] = inv;
}

template <int C>
__global__ void weisz_edge_k(const float* __restrict__ h, const float* __restrict__ med,
                             const int* __restrict__ src, const int* __restrict__ dst,
                             float* __restrict__ ws, float* __restrict__ wt, int E) {
  constexpr int LPE = (C <= 16) ? 16 : 32;
  int gid = blockIdx.x * blockDim.x + threadIdx.x;
  int e = gid / LPE, lane = gid % LPE;
  if (e >= E) return;
  int s = src[e], t = dst[e];
  float m0 = h[s * C + lane], mu0 = med[t * C + lane];
  float d0 = m0 - mu0, sq = d0 * d0;
  float m1 = 0.0f;
  if (C > 32 && lane < C - 32) {
    m1 = h[s * C + lane + 32];
    float d1 = m1 - med[t * C + lane + 32];
    sq += d1 * d1;
  }
  sq = grp_sum<LPE>(sq);
  float inv = 1.0f / sqrtf(fmaxf(sq, EPS2V));
  atomicAdd(ws + t * C + lane, m0 * inv);
  if (C > 32 && lane < C - 32) atomicAdd(ws + t * C + lane + 32, m1 * inv);
  if (lane == 0) atomicAdd(wt + t, inv);
}

template <int C>
__global__ void update_k(const float* __restrict__ ws, const float* __restrict__ wt,
                         float* __restrict__ med, int* __restrict__ act, int Nn) {
  constexpr int LPE = (C <= 16) ? 16 : 32;
  int gid = blockIdx.x * blockDim.x + threadIdx.x;
  int i = gid / LPE, lane = gid % LPE;
  if (i >= Nn) return;
  float invw = 1.0f / wt[i];
  float n0 = ws[i * C + lane] * invw;
  float df0 = n0 - med[i * C + lane];
  float sq = df0 * df0;
  float n1 = 0.0f;
  if (C > 32 && lane < C - 32) {
    n1 = ws[i * C + lane + 32] * invw;
    float df1 = n1 - med[i * C + lane + 32];
    sq += df1 * df1;
  }
  sq = grp_sum<LPE>(sq);
  float delta = sqrtf(sq);
  bool upd = (act[i] != 0) && (delta >= EPSV);   // freeze logic matches reference
  if (upd) {
    med[i * C + lane] = n0;
    if (C > 32 && lane < C - 32) med[i * C + lane + 32] = n1;
  }
  if (lane == 0) act[i] = upd ? 1 : 0;
}

// relu + log_softmax over 40 classes; one wave per node.
__global__ void logsoftmax_k(const float* __restrict__ med, float* __restrict__ out, int Nn) {
  int gid = blockIdx.x * blockDim.x + threadIdx.x;
  int i = gid >> 5, lane = gid & 31;
  if (i >= Nn) return;
  bool has1 = lane < 8;
  float v0 = fmaxf(med[i * 40 + lane], 0.0f);
  float v1 = has1 ? fmaxf(med[i * 40 + 32 + lane], 0.0f) : -INFINITY;
  float mx = grp_max32(fmaxf(v0, v1));
  float s = expf(v0 - mx) + (has1 ? expf(v1 - mx) : 0.0f);
  s = grp_sum<32>(s);
  float lse = mx + logf(s);
  out[i * 40 + lane] = v0 - lse;
  if (has1) out[i * 40 + 32 + lane] = v1 - lse;
}

// ---------------------------------------------------------------- launch ----
extern "C" void kernel_launch(void* const* d_in, const int* in_sizes, int n_in,
                              void* d_out, int out_size, void* d_ws, size_t ws_size,
                              hipStream_t stream) {
  (void)n_in; (void)out_size; (void)ws_size;
  const float* x  = (const float*)d_in[0];
  const float* W1 = (const float*)d_in[1];
  const float* b1 = (const float*)d_in[2];
  const float* W2 = (const float*)d_in[3];
  const float* b2 = (const float*)d_in[4];
  const int*   ei = (const int*)d_in[5];

  const int IN_F = 128, NC = 40, ITERS = 10;
  const int Nn = in_sizes[0] / IN_F;
  const int E  = in_sizes[5] / 2;
  const int* src = ei;
  const int* dst = ei + E;

  // workspace layout (~70 MB; entirely L2-resident on MI455X)
  char* base = (char*)d_ws;
  size_t off = 0;
  auto take = [&](size_t bytes) -> void* {
    void* p = base + off;
    off = (off + bytes + 255) & ~(size_t)255;
    return p;
  };
  float* H1  = (float*)take((size_t)Nn * 16 * 4);  // layer1 messages
  float* H1P = (float*)take((size_t)Nn * 32 * 4);  // relu(med1) zero-padded K=32
  float* MED = (float*)take((size_t)Nn * NC * 4);
  float* WS  = (float*)take((size_t)Nn * NC * 4);
  float* H2  = (float*)take((size_t)Nn * NC * 4);
  float* WT  = (float*)take((size_t)Nn * 4);
  float* CNT = (float*)take((size_t)Nn * 4);
  int*   ACT = (int*)take((size_t)Nn * 4);
  float* W2P = (float*)take((size_t)48 * 32 * 4);  // W2 zero-padded [48,32]
  float* B2P = (float*)take((size_t)48 * 4);

  const int BLK = 256;
  auto blocks = [](long long t, int b) { return (unsigned)((t + b - 1) / b); };
  const unsigned mt = (unsigned)(((Nn + 15) / 16 + 3) / 4);  // 4 M-tiles / block

  // ---- layer 1: h1 = x @ W1^T + b1 (WMMA K=128, 4 wmma unrolled) ----
  gemm_wmma_f16<128><<<dim3(mt, 1), 128, 0, stream>>>(x, W1, b1, H1, Nn, 16);

  mean_init_k<16><<<blocks((long long)Nn * 16, BLK), BLK, 0, stream>>>(H1, WS, CNT, Nn);
  mean_edge_k<16><<<blocks((long long)E * 16, BLK), BLK, 0, stream>>>(H1, src, dst, WS, CNT, E);
  mean_final_k<16><<<blocks((long long)Nn * 16, BLK), BLK, 0, stream>>>(WS, CNT, MED, ACT, Nn);
  for (int it = 0; it < ITERS; ++it) {
    self_init_k<16><<<blocks((long long)Nn * 16, BLK), BLK, 0, stream>>>(H1, MED, WS, WT, Nn);
    weisz_edge_k<16><<<blocks((long long)E * 16, BLK), BLK, 0, stream>>>(H1, MED, src, dst, WS, WT, E);
    update_k<16><<<blocks((long long)Nn * 16, BLK), BLK, 0, stream>>>(WS, WT, MED, ACT, Nn);
  }

  // ---- layer 2 prep: pad operands, relu(med1) -> [N,32] ----
  pad_w2_k<<<blocks(48 * 32, BLK), BLK, 0, stream>>>(W2, b2, W2P, B2P);
  relu_pad_k<<<blocks((long long)Nn * 32, BLK), BLK, 0, stream>>>(MED, H1P, Nn);

  // ---- layer 2: h2 = relu(med1) @ W2^T + b2 (WMMA K=32, guard-free loads) ----
  gemm_wmma_f16<32><<<dim3(mt, 3), 128, 0, stream>>>(H1P, W2P, B2P, H2, Nn, NC);

  mean_init_k<40><<<blocks((long long)Nn * 40, BLK), BLK, 0, stream>>>(H2, WS, CNT, Nn);
  mean_edge_k<40><<<blocks((long long)E * 32, BLK), BLK, 0, stream>>>(H2, src, dst, WS, CNT, E);
  mean_final_k<40><<<blocks((long long)Nn * 40, BLK), BLK, 0, stream>>>(WS, CNT, MED, ACT, Nn);
  for (int it = 0; it < ITERS; ++it) {
    self_init_k<40><<<blocks((long long)Nn * 32, BLK), BLK, 0, stream>>>(H2, MED, WS, WT, Nn);
    weisz_edge_k<40><<<blocks((long long)E * 32, BLK), BLK, 0, stream>>>(H2, MED, src, dst, WS, WT, E);
    update_k<40><<<blocks((long long)Nn * 32, BLK), BLK, 0, stream>>>(WS, WT, MED, ACT, Nn);
  }

  // ---- final: log_softmax(relu(med2)) -> d_out ----
  logsoftmax_k<<<blocks((long long)Nn * 32, BLK), BLK, 0, stream>>>(MED, (float*)d_out, Nn);
}
